// InvResMLP_24850680774735
// MI455X (gfx1250) — matrix operand order repaired and checked by reading, not compiled, not run on hardware
//
#include <hip/hip_runtime.h>

typedef __bf16 bf16;
typedef __attribute__((ext_vector_type(16))) __bf16 v16bf;
typedef __attribute__((ext_vector_type(8)))  __bf16 v8bf;
typedef __attribute__((ext_vector_type(8)))  float  v8f;
typedef __attribute__((ext_vector_type(4)))  unsigned int u32x4;
typedef __attribute__((ext_vector_type(8)))  int i32x8;
typedef __attribute__((ext_vector_type(4)))  int i32x4;

#define BB     8
#define NN     4096
#define CC     128
#define KNN    32
#define CINP   160            // 128 feat + 3 rel + 29 pad (5 K-tiles of 32)
#define HID    512
#define RADIUS 0.15f
#define R2     (RADIUS * RADIUS)
#define INVR   (1.0f / RADIUS)
#define EPS    1e-5f
#define NPTS   (BB * NN)
#define POSOFF ((size_t)NPTS * 3)

// swizzled weight sizes (fragment-major: [kt][nt][lane][e])
#define SB1_ELEMS (5 * 8 * 512)     // 20480
#define SB2_ELEMS (4 * 32 * 512)    // 65536
#define SB3_ELEMS (16 * 8 * 512)    // 65536

// ---- workspace layout (bytes) ----
#define WS_IDX   0                                      // u16 [NPTS*KNN]
#define WS_XBF   (WS_IDX  + (size_t)NPTS*KNN*2)         // bf16 [NPTS*CC]
#define WS_FBUF  (WS_XBF  + (size_t)NPTS*CC*2)          // bf16 [NPTS*CC]
#define WS_B1    (WS_FBUF + (size_t)NPTS*CC*2)          // bf16 [SB1]
#define WS_B2    (WS_B1   + (size_t)SB1_ELEMS*2)        // bf16 [SB2]
#define WS_B3    (WS_B2   + (size_t)SB2_ELEMS*2)        // bf16 [SB3]
#define WS_S1    (WS_B3   + (size_t)SB3_ELEMS*2)
#define WS_BB1   (WS_S1   + (size_t)CC*4)
#define WS_S2    (WS_BB1  + (size_t)CC*4)
#define WS_BB2   (WS_S2   + (size_t)HID*4)
#define WS_S3    (WS_BB2  + (size_t)HID*4)
#define WS_BB3   (WS_S3   + (size_t)CC*4)

union Frag { v16bf v; v8bf h[2]; };

__device__ __forceinline__ float xor16_max(float v) {
    int s = __builtin_amdgcn_ds_swizzle(__float_as_int(v), 0x401f); // xor=16,and=0x1f
    return fmaxf(v, __int_as_float(s));
}

// ---------------------------------------------------------------------------
// TDM gather: 16 rows (16-bit indices) x 128 bf16 from a (NN x 128) bf16
// tensor into LDS, padded +32 bf16 per row so rows land at CINP stride.
// D# per CDNA5 ISA (gather_mode, pad_enable).
// ---------------------------------------------------------------------------
__device__ __forceinline__ void tdm_gather16(unsigned lds_addr,
                                             unsigned long long gaddr,
                                             const int* gi) {
    u32x4 g0 = {0x80000001u,                                   // count=1, gather_mode=1, idx16
                lds_addr,                                      // lds_addr
                (unsigned)gaddr,                               // global_addr[31:0]
                0x80000000u | (unsigned)((gaddr >> 32) & 0x1FFFFFFull)}; // [56:32] | type=2
    i32x8 g1 = {(int)0x1F510000,  // data_size=2B, pad_en, pad_interval=64DW, pad_amount=16DW
                (int)(CC << 16),  // tensor_dim0 = 128 (low 16 in [31:16])
                (int)(NN << 16),  // tensor_dim0 hi=0 | tensor_dim1 = 4096 (low 16)
                (int)(CC << 16),  // tensor_dim1 hi=0 | tile_dim0 = 128
                16,               // tile_dim1 = 16 valid indices (tile_dim2 ignored)
                CC,               // tensor_dim0_stride = 128
                0, 0};
    i32x4 g2 = {gi[0], gi[1], gi[2], gi[3]};   // row_index_0..7  (u16 pairs)
    i32x4 g3 = {gi[4], gi[5], gi[6], gi[7]};   // row_index_8..15
#if defined(__clang_major__) && __clang_major__ >= 23
    i32x8 g4 = {0, 0, 0, 0, 0, 0, 0, 0};
    __builtin_amdgcn_tensor_load_to_lds(g0, g1, g2, g3, g4, 0);
#else
    __builtin_amdgcn_tensor_load_to_lds(g0, g1, g2, g3, 0);
#endif
}

// ---------------------------------------------------------------------------
// K0a: repack weights into fragment-major bf16 (lane-contiguous 32B chunks),
// fold BN into per-channel scale/bias.
// ---------------------------------------------------------------------------
__global__ void prep_kernel(const float* __restrict__ W1, const float* __restrict__ g1,
                            const float* __restrict__ b1, const float* __restrict__ m1,
                            const float* __restrict__ v1,
                            const float* __restrict__ W2, const float* __restrict__ g2,
                            const float* __restrict__ b2, const float* __restrict__ m2,
                            const float* __restrict__ v2,
                            const float* __restrict__ W3, const float* __restrict__ g3,
                            const float* __restrict__ b3, const float* __restrict__ m3,
                            const float* __restrict__ v3,
                            bf16* SwB1, bf16* SwB2, bf16* SwB3,
                            float* s1, float* bb1, float* s2, float* bb2,
                            float* s3, float* bb3) {
    int t = blockIdx.x * blockDim.x + threadIdx.x;
    if (t < SB1_ELEMS) {               // W1: K in [0,160) (feat|rel|pad), N in [0,128), NT=8
        int tile = t >> 9, r = t & 511, lane = r >> 4, e = r & 15;
        int kt = tile / 8, nt = tile % 8;
        int K = kt * 32 + (lane >> 4) * 8 + (e & 7) + (e >> 3) * 16, Nc = nt * 16 + (lane & 15);
        float w = 0.0f;
        if (K < CC)          w = W1[Nc * (CC + 3) + 3 + K];
        else if (K < CC + 3) w = W1[Nc * (CC + 3) + (K - CC)];
        SwB1[t] = (bf16)w;
        return;
    }
    t -= SB1_ELEMS;
    if (t < SB2_ELEMS) {               // W2 (512x128): K in [0,128), N in [0,512), NT=32
        int tile = t >> 9, r = t & 511, lane = r >> 4, e = r & 15;
        int kt = tile / 32, nt = tile % 32;
        int K = kt * 32 + (lane >> 4) * 8 + (e & 7) + (e >> 3) * 16, Nc = nt * 16 + (lane & 15);
        SwB2[t] = (bf16)W2[(size_t)Nc * CC + K];
        return;
    }
    t -= SB2_ELEMS;
    if (t < SB3_ELEMS) {               // W3 (128x512): K in [0,512), N in [0,128), NT=8
        int tile = t >> 9, r = t & 511, lane = r >> 4, e = r & 15;
        int kt = tile / 8, nt = tile % 8;
        int K = kt * 32 + (lane >> 4) * 8 + (e & 7) + (e >> 3) * 16, Nc = nt * 16 + (lane & 15);
        SwB3[t] = (bf16)W3[(size_t)Nc * HID + K];
        return;
    }
    t -= SB3_ELEMS;
    if (t < CC)  { float s = g1[t] * rsqrtf(v1[t] + EPS); s1[t] = s; bb1[t] = b1[t] - m1[t] * s; return; }
    t -= CC;
    if (t < HID) { float s = g2[t] * rsqrtf(v2[t] + EPS); s2[t] = s; bb2[t] = b2[t] - m2[t] * s; return; }
    t -= HID;
    if (t < CC)  { float s = g3[t] * rsqrtf(v3[t] + EPS); s3[t] = s; bb3[t] = b3[t] - m3[t] * s; return; }
}

// K0b: x (f32) -> xbf (bf16) so TDM can gather matrix-ready rows
__global__ void xcast_kernel(const float* __restrict__ x, bf16* __restrict__ xbf) {
    int t = blockIdx.x * blockDim.x + threadIdx.x;
    if (t < NPTS * CC) xbf[t] = (bf16)x[t];
}

__global__ void copy_pos_kernel(const float* __restrict__ pos, float* __restrict__ out) {
    int t = blockIdx.x * blockDim.x + threadIdx.x;
    if (t < (int)POSOFF) out[t] = pos[t];
}

// ---------------------------------------------------------------------------
// K1: ball query — pos[b] staged SoA in LDS once per 16-centroid block,
// then wave32 ballot streaming scan for the first KNN hits in index order.
// ---------------------------------------------------------------------------
#define BQW 16
__global__ __launch_bounds__(512) void ballq_kernel(const float* __restrict__ pos,
                                                    unsigned short* __restrict__ idx) {
    __shared__ float px[NN], py[NN], pz[NN];          // 48 KB
    int b = blockIdx.x >> 8;                          // 256 blocks per batch
    int m0 = (blockIdx.x & 255) * BQW;
    const float* pb = pos + (size_t)b * NN * 3;
    for (int i = threadIdx.x; i < NN; i += 512) {
        px[i] = pb[i * 3 + 0];
        py[i] = pb[i * 3 + 1];
        pz[i] = pb[i * 3 + 2];
    }
    __syncthreads();
    int lane = threadIdx.x & 31, w = threadIdx.x >> 5;
    int m = m0 + w;
    float cx = px[m], cy = py[m], cz = pz[m];
    unsigned short* outp = idx + (size_t)(b * NN + m) * KNN;
    int cnt = 0, first = -1;
    for (int j0 = 0; j0 < NN && cnt < KNN; j0 += 32) {
        int j = j0 + lane;
        float dx = px[j] - cx, dy = py[j] - cy, dz = pz[j] - cz;
        bool hit = (dx * dx + dy * dy + dz * dz) < R2;
        unsigned mask = __builtin_amdgcn_ballot_w32(hit);
        if (first < 0 && mask) first = j0 + __builtin_ctz(mask);
        if (hit) {
            int p = cnt + __builtin_popcount(mask & ((1u << lane) - 1u));
            if (p < KNN) outp[p] = (unsigned short)j;
        }
        cnt += __builtin_popcount(mask);
    }
    if (first < 0) first = 0;
    if (lane >= cnt && lane < KNN) outp[lane] = (unsigned short)first;
}

// ---------------------------------------------------------------------------
// K2: TDM-gathered conv1 (32x160 . 160x128 bf16 WMMA) + k-max + BN1 + ReLU.
// One wave per point; features DMA'd into LDS by two tensor_load_to_lds
// gathers; rel-coords/pad written by VALU into the padded columns.
// ---------------------------------------------------------------------------
#define W2V 2
__global__ __launch_bounds__(64) void conv1_kernel(const float* __restrict__ pos,
                                                   const bf16* __restrict__ xbf,
                                                   const unsigned short* __restrict__ idx,
                                                   const bf16* __restrict__ SwB1,
                                                   const float* __restrict__ s1,
                                                   const float* __restrict__ bb1,
                                                   bf16* __restrict__ fbuf) {
    __shared__ bf16 Atile[W2V][KNN * CINP];           // 2 x 10 KB
    int lane = threadIdx.x & 31;
    int w    = threadIdx.x >> 5;
    int pt   = blockIdx.x * W2V + w;
    int b = pt >> 12, m = pt & (NN - 1);
    const float* pb = pos + (size_t)b * NN * 3;

    __builtin_prefetch(SwB1 + (size_t)lane * 640, 0, 1);

    const unsigned short* myidx = idx + (size_t)pt * KNN;
    int nbr = myidx[lane];
    bf16* At = Atile[w];

    // two async TDM gathers: rows 0..15 and 16..31 of the A tile
    {
        unsigned lds0 = (unsigned)(unsigned long long)(void*)At;
        unsigned long long ga = (unsigned long long)(const void*)(xbf + (size_t)b * NN * CC);
        const int* gi = (const int*)myidx;
        tdm_gather16(lds0, ga, gi);
        tdm_gather16(lds0 + 16 * CINP * 2, ga, gi + 8);
    }
    // rel coords + zero pad into columns 128..159 (disjoint from TDM bytes)
    {
        float cx = pb[m * 3 + 0], cy = pb[m * 3 + 1], cz = pb[m * 3 + 2];
        int base = lane * CINP + CC;
        At[base + 0] = (bf16)((pb[nbr * 3 + 0] - cx) * INVR);
        At[base + 1] = (bf16)((pb[nbr * 3 + 1] - cy) * INVR);
        At[base + 2] = (bf16)((pb[nbr * 3 + 2] - cz) * INVR);
        for (int d = 3; d < CINP - CC; ++d) At[base + d] = (bf16)0.0f;
    }
    __builtin_amdgcn_s_wait_tensorcnt(0);
    __syncthreads();

    int half = lane >> 4, mc = lane & 15;
    v8f zero = {0.f, 0.f, 0.f, 0.f, 0.f, 0.f, 0.f, 0.f};
    v8f acc[2][8];
#pragma unroll
    for (int mt = 0; mt < 2; ++mt)
#pragma unroll
        for (int nt = 0; nt < 8; ++nt) acc[mt][nt] = zero;

#pragma unroll
    for (int kt = 0; kt < 5; ++kt) {
        Frag a0, a1;
        const bf16* ap0 = At + mc * CINP + kt * 32 + half * 8;
        const bf16* ap1 = At + (16 + mc) * CINP + kt * 32 + half * 8;
        a0.h[0] = *(const v8bf*)ap0;  a0.h[1] = *(const v8bf*)(ap0 + 16);
        a1.h[0] = *(const v8bf*)ap1;  a1.h[1] = *(const v8bf*)(ap1 + 16);
#pragma unroll
        for (int nt = 0; nt < 8; ++nt) {
            Frag bfr;
            const v8bf* bp = (const v8bf*)SwB1 + ((size_t)((kt * 8 + nt) * 32 + lane)) * 2;
            bfr.h[0] = bp[0];  bfr.h[1] = bp[1];
            acc[0][nt] = __builtin_amdgcn_wmma_f32_16x16x32_bf16(
                false, a0.v, false, bfr.v, (short)0, acc[0][nt], false, false);
            acc[1][nt] = __builtin_amdgcn_wmma_f32_16x16x32_bf16(
                false, a1.v, false, bfr.v, (short)0, acc[1][nt], false, false);
        }
    }

    // max over k (rows) then BN1+ReLU (scale>0 commutes with max)
#pragma unroll
    for (int nt = 0; nt < 8; ++nt) {
        float mx = acc[0][nt][0];
#pragma unroll
        for (int v = 1; v < 8; ++v) mx = fmaxf(mx, acc[0][nt][v]);
#pragma unroll
        for (int v = 0; v < 8; ++v) mx = fmaxf(mx, acc[1][nt][v]);
        mx = xor16_max(mx);
        int ch = nt * 16 + mc;
        float val = fmaxf(mx * s1[ch] + bb1[ch], 0.0f);
        if (half == 0) fbuf[(size_t)pt * CC + ch] = (bf16)val;
    }
}

// ---------------------------------------------------------------------------
// K3: fused conv2+BN2+ReLU+conv3+BN3+residual+ReLU, 16-point tile, 4 waves.
// ---------------------------------------------------------------------------
__global__ __launch_bounds__(128) void mlp_kernel(const bf16* __restrict__ fbuf,
                                                  const bf16* __restrict__ SwB2,
                                                  const bf16* __restrict__ SwB3,
                                                  const float* __restrict__ s2,
                                                  const float* __restrict__ bb2,
                                                  const float* __restrict__ s3,
                                                  const float* __restrict__ bb3,
                                                  const float* __restrict__ x,
                                                  float* __restrict__ out) {
    __shared__ bf16 h1[16 * HID];                     // 16 KB
    int lane = threadIdx.x & 31;
    int w    = threadIdx.x >> 5;
    int p0   = blockIdx.x * 16;
    int half = lane >> 4, mc = lane & 15;

    __builtin_prefetch(SwB2 + (size_t)threadIdx.x * 512, 0, 1);
    __builtin_prefetch(SwB3 + (size_t)threadIdx.x * 512, 0, 1);

    v8f zero = {0.f, 0.f, 0.f, 0.f, 0.f, 0.f, 0.f, 0.f};
    v8f acc[8];
#pragma unroll
    for (int nt = 0; nt < 8; ++nt) acc[nt] = zero;

    // ---- stage 1: h = f(16x128) . W2 (128x512); this wave: cols [w*128, +128)
#pragma unroll
    for (int kt = 0; kt < 4; ++kt) {
        Frag a;
        const bf16* ap = fbuf + (size_t)(p0 + mc) * CC + kt * 32 + half * 8;
        a.h[0] = *(const v8bf*)ap;  a.h[1] = *(const v8bf*)(ap + 16);
#pragma unroll
        for (int nt = 0; nt < 8; ++nt) {
            Frag bfr;
            const v8bf* bp = (const v8bf*)SwB2 +
                             ((size_t)((kt * 32 + (w * 8 + nt)) * 32 + lane)) * 2;
            bfr.h[0] = bp[0];  bfr.h[1] = bp[1];
            acc[nt] = __builtin_amdgcn_wmma_f32_16x16x32_bf16(
                false, a.v, false, bfr.v, (short)0, acc[nt], false, false);
        }
    }
    // BN2 + ReLU -> LDS bf16
#pragma unroll
    for (int nt = 0; nt < 8; ++nt) {
        int ch = w * 128 + nt * 16 + mc;
        float s = s2[ch], bb = bb2[ch];
#pragma unroll
        for (int v = 0; v < 8; ++v) {
            int mrow = half * 8 + v;
            h1[mrow * HID + ch] = (bf16)fmaxf(acc[nt][v] * s + bb, 0.0f);
        }
    }
    __syncthreads();

    // ---- stage 2: o = h1(16x512) . W3 (512x128); this wave: cols [w*32, +32)
    v8f acc2[2] = {zero, zero};
#pragma unroll
    for (int kt = 0; kt < 16; ++kt) {
        Frag a;
        const bf16* ap = h1 + mc * HID + kt * 32 + half * 8;
        a.h[0] = *(const v8bf*)ap;  a.h[1] = *(const v8bf*)(ap + 16);
#pragma unroll
        for (int ct = 0; ct < 2; ++ct) {
            Frag bfr;
            const v8bf* bp = (const v8bf*)SwB3 +
                             ((size_t)((kt * 8 + (w * 2 + ct)) * 32 + lane)) * 2;
            bfr.h[0] = bp[0];  bfr.h[1] = bp[1];
            acc2[ct] = __builtin_amdgcn_wmma_f32_16x16x32_bf16(
                false, a.v, false, bfr.v, (short)0, acc2[ct], false, false);
        }
    }
    // BN3 + residual + ReLU -> out
    float* outx = out + POSOFF;
#pragma unroll
    for (int ct = 0; ct < 2; ++ct) {
        int ch = w * 32 + ct * 16 + mc;
        float s = s3[ch], bb = bb3[ch];
#pragma unroll
        for (int v = 0; v < 8; ++v) {
            int p = p0 + half * 8 + v;
            float h = acc2[ct][v] * s + bb;
            outx[(size_t)p * CC + ch] = fmaxf(x[(size_t)p * CC + ch] + h, 0.0f);
        }
    }
}

// ---------------------------------------------------------------------------
extern "C" void kernel_launch(void* const* d_in, const int* in_sizes, int n_in,
                              void* d_out, int out_size, void* d_ws, size_t ws_size,
                              hipStream_t stream) {
    const float* pos = (const float*)d_in[0];
    const float* x   = (const float*)d_in[1];
    const float* W1  = (const float*)d_in[2];
    const float* g1  = (const float*)d_in[3];
    const float* b1  = (const float*)d_in[4];
    const float* m1  = (const float*)d_in[5];
    const float* v1  = (const float*)d_in[6];
    const float* W2  = (const float*)d_in[7];
    const float* g2  = (const float*)d_in[8];
    const float* b2  = (const float*)d_in[9];
    const float* m2  = (const float*)d_in[10];
    const float* v2  = (const float*)d_in[11];
    const float* W3  = (const float*)d_in[12];
    const float* g3  = (const float*)d_in[13];
    const float* b3  = (const float*)d_in[14];
    const float* m3  = (const float*)d_in[15];
    const float* v3  = (const float*)d_in[16];
    float* out = (float*)d_out;

    char* ws = (char*)d_ws;
    unsigned short* idx = (unsigned short*)(ws + WS_IDX);
    bf16*  xbf  = (bf16*)(ws + WS_XBF);
    bf16*  fbuf = (bf16*)(ws + WS_FBUF);
    bf16*  SwB1 = (bf16*)(ws + WS_B1);
    bf16*  SwB2 = (bf16*)(ws + WS_B2);
    bf16*  SwB3 = (bf16*)(ws + WS_B3);
    float* s1   = (float*)(ws + WS_S1);
    float* bb1  = (float*)(ws + WS_BB1);
    float* s2   = (float*)(ws + WS_S2);
    float* bb2  = (float*)(ws + WS_BB2);
    float* s3   = (float*)(ws + WS_S3);
    float* bb3  = (float*)(ws + WS_BB3);

    int prep_elems = SB1_ELEMS + SB2_ELEMS + SB3_ELEMS + CC + HID + CC;
    prep_kernel<<<(prep_elems + 255) / 256, 256, 0, stream>>>(
        W1, g1, b1, m1, v1, W2, g2, b2, m2, v2, W3, g3, b3, m3, v3,
        SwB1, SwB2, SwB3, s1, bb1, s2, bb2, s3, bb3);

    xcast_kernel<<<(NPTS * CC + 255) / 256, 256, 0, stream>>>(x, xbf);

    copy_pos_kernel<<<((int)POSOFF + 255) / 256, 256, 0, stream>>>(pos, out);

    ballq_kernel<<<NPTS / BQW, 512, 0, stream>>>(pos, idx);

    conv1_kernel<<<NPTS / W2V, 64, 0, stream>>>(pos, xbf, idx, SwB1, s1, bb1, fbuf);

    mlp_kernel<<<NPTS / 16, 128, 0, stream>>>(fbuf, SwB2, SwB3, s2, bb2, s3, bb3, x, out);
}
